// DecoderStepForExport_22290880266263
// MI455X (gfx1250) — matrix operand order repaired and checked by reading, not compiled, not run on hardware
//
#include <hip/hip_runtime.h>
#include <math.h>

// ---------------- problem sizes ----------------
static constexpr int B = 32, L = 2048, S = 256;
static constexpr int D = 1024, E = 1024, EMB = 256, V = 4096, T = 64;
static constexpr int XK = EMB + E;           // 1280
static constexpr float NEGV = -1000000000.0f;

// ---------------- WMMA types ----------------
typedef __attribute__((ext_vector_type(16))) __bf16 v16bf;
typedef __attribute__((ext_vector_type(8)))  float  v8f;

// ---------------- workspace layout (floats) ----------------
static constexpr size_t WS_X      = 0;                      // B*XK
static constexpr size_t WS_CTXQ   = WS_X      + B * XK;     // B*E
static constexpr size_t WS_SCORES = WS_CTXQ   + B * E;      // B*L
static constexpr size_t WS_STATS  = WS_SCORES + B * L;      // 2*B
static constexpr size_t WS_PART   = WS_STATS  + 2 * B;      // B*8*E
static constexpr size_t WS_GI     = WS_PART   + B * 8 * E;  // B*3*D
static constexpr size_t WS_GH     = WS_GI     + B * 3 * D;  // B*3*D
static constexpr size_t WS_QSPAN  = WS_GH     + B * 3 * D;  // 4*B*E
static constexpr size_t WS_QREF   = WS_QSPAN  + 4 * B * E;  // 2*B*D
static constexpr size_t WS_QK     = WS_QREF   + 2 * B * D;  // 2*B*D
static constexpr size_t WS_CB     = WS_QK     + 2 * B * D;  // 2*B
static constexpr size_t WS_WT     = WS_CB     + 2 * B;      // 2*D*E (transposed r{0,1}k_w)

// ---------------- output layout (floats) ----------------
static constexpr size_t O_HID = 0;                    // B*D
static constexpr size_t O_OP  = O_HID + B * D;        // B*V
static constexpr size_t O_A0  = O_OP  + B * V;        // B*T
static constexpr size_t O_A1  = O_A0  + B * T;        // B*T
static constexpr size_t O_S0S = O_A1  + B * T;        // B*L
static constexpr size_t O_S0E = O_S0S + B * L;
static constexpr size_t O_S1S = O_S0E + B * L;
static constexpr size_t O_S1E = O_S1S + B * L;
static constexpr size_t O_R0  = O_S1E + B * L;        // B*S
static constexpr size_t O_R1  = O_R0  + B * S;

// ---------------- helpers ----------------
__device__ __forceinline__ float wave_sum(float v) {
  #pragma unroll
  for (int o = 16; o > 0; o >>= 1) v += __shfl_xor(v, o, 32);
  return v;
}
__device__ __forceinline__ float wave_max(float v) {
  #pragma unroll
  for (int o = 16; o > 0; o >>= 1) v = fmaxf(v, __shfl_xor(v, o, 32));
  return v;
}

// A-fragment (16x32 bf16): lane half h, row r. p = &A[row][kb] + h*8.
// elements 0..7  = K(h*8 .. h*8+7), elements 8..15 = K(16+h*8 .. 16+h*8+7)
__device__ __forceinline__ v16bf load_frag_a(const float* __restrict__ p) {
  float4 x0 = *(const float4*)(p);
  float4 x1 = *(const float4*)(p + 4);
  float4 x2 = *(const float4*)(p + 16);
  float4 x3 = *(const float4*)(p + 20);
  v16bf f;
  f[0]  = (__bf16)x0.x; f[1]  = (__bf16)x0.y; f[2]  = (__bf16)x0.z; f[3]  = (__bf16)x0.w;
  f[4]  = (__bf16)x1.x; f[5]  = (__bf16)x1.y; f[6]  = (__bf16)x1.z; f[7]  = (__bf16)x1.w;
  f[8]  = (__bf16)x2.x; f[9]  = (__bf16)x2.y; f[10] = (__bf16)x2.z; f[11] = (__bf16)x2.w;
  f[12] = (__bf16)x3.x; f[13] = (__bf16)x3.y; f[14] = (__bf16)x3.z; f[15] = (__bf16)x3.w;
  return f;
}

// B-fragment (32x16 bf16): lane half h, col r. p = &W[col][kb] + h*8.
// elements 0..7  = K(h*8 .. h*8+7), elements 8..15 = K(16+h*8 .. 16+h*8+7)
__device__ __forceinline__ v16bf load_frag_b(const float* __restrict__ p) {
  float4 x0 = *(const float4*)(p);
  float4 x1 = *(const float4*)(p + 4);
  float4 x2 = *(const float4*)(p + 16);
  float4 x3 = *(const float4*)(p + 20);
  v16bf f;
  f[0]  = (__bf16)x0.x; f[1]  = (__bf16)x0.y; f[2]  = (__bf16)x0.z; f[3]  = (__bf16)x0.w;
  f[4]  = (__bf16)x1.x; f[5]  = (__bf16)x1.y; f[6]  = (__bf16)x1.z; f[7]  = (__bf16)x1.w;
  f[8]  = (__bf16)x2.x; f[9]  = (__bf16)x2.y; f[10] = (__bf16)x2.z; f[11] = (__bf16)x2.w;
  f[12] = (__bf16)x3.x; f[13] = (__bf16)x3.y; f[14] = (__bf16)x3.z; f[15] = (__bf16)x3.w;
  return f;
}

// =====================================================================
// bf16 WMMA GEMM: C(M,N) = A(M,K) @ W^T + bias, W is (N,K) row-major.
// Block = 256 thr = 8 waves as 2(M)x4(N); wave tile = 32x64; block tile 64x256.
// K must be a multiple of 32.
// =====================================================================
__global__ void gemm_bf16(const float* __restrict__ A, const float* __restrict__ W,
                          const float* __restrict__ bias, float* __restrict__ C,
                          int M, int N, int K) {
  const int lane = threadIdx.x & 31;
  const int wave = threadIdx.x >> 5;
  const int wm = wave >> 2, wn = wave & 3;
  const int mBase = blockIdx.x * 64 + wm * 32;
  const int nBase = blockIdx.y * 256 + wn * 64;
  const int r = lane & 15, h = lane >> 4;

  int rowA0 = mBase + r;      if (rowA0 > M - 1) rowA0 = M - 1;
  int rowA1 = mBase + 16 + r; if (rowA1 > M - 1) rowA1 = M - 1;
  const float* a0p = A + (size_t)rowA0 * K + h * 8;
  const float* a1p = A + (size_t)rowA1 * K + h * 8;
  const float* bp[4];
  #pragma unroll
  for (int u = 0; u < 4; ++u) {
    int col = nBase + u * 16 + r; if (col > N - 1) col = N - 1;
    bp[u] = W + (size_t)col * K + h * 8;
  }

  v8f acc[2][4] = {};

  for (int kb = 0; kb < K; kb += 32) {
    v16bf a0 = load_frag_a(a0p + kb);
    v16bf a1 = load_frag_a(a1p + kb);
    #pragma unroll
    for (int u = 0; u < 4; ++u) {
      v16bf b = load_frag_b(bp[u] + kb);
      acc[0][u] = __builtin_amdgcn_wmma_f32_16x16x32_bf16(
          false, a0, false, b, (short)0, acc[0][u], false, false);
      acc[1][u] = __builtin_amdgcn_wmma_f32_16x16x32_bf16(
          false, a1, false, b, (short)0, acc[1][u], false, false);
    }
  }

  #pragma unroll
  for (int t = 0; t < 2; ++t) {
    #pragma unroll
    for (int u = 0; u < 4; ++u) {
      int n = nBase + u * 16 + r;
      if (n < N) {
        float bv = bias ? bias[n] : 0.0f;
        #pragma unroll
        for (int i = 0; i < 8; ++i) {
          int m = mBase + t * 16 + h * 8 + i;
          if (m < M) C[(size_t)m * N + n] = acc[t][u][i] + bv;
        }
      }
    }
  }
}

// =====================================================================
// 32x32 LDS-tiled transpose: dst(Cn,R) = src(R,Cn)^T
// =====================================================================
__global__ void k_transpose(const float* __restrict__ src, float* __restrict__ dst,
                            int R, int Cn) {
  __shared__ float tile[32][33];
  int c0 = blockIdx.x * 32, r0 = blockIdx.y * 32;
  int tx = threadIdx.x & 31, ty = threadIdx.x >> 5;   // 256 thr = 32x8
  #pragma unroll
  for (int i = ty; i < 32; i += 8)
    tile[i][tx] = src[(size_t)(r0 + i) * Cn + c0 + tx];
  __syncthreads();
  #pragma unroll
  for (int i = ty; i < 32; i += 8)
    dst[(size_t)(c0 + i) * R + r0 + tx] = tile[tx][i];
}

// =====================================================================
// elementwise / reduction kernels
// =====================================================================

__global__ void k_gather(const int* __restrict__ ids, const float* __restrict__ emb,
                         float* __restrict__ x) {
  int b = blockIdx.x;
  x[(size_t)b * XK + threadIdx.x] = emb[(size_t)ids[b] * EMB + threadIdx.x];
}

__global__ void k_scores(const float* __restrict__ ctxq, const float* __restrict__ enc,
                         const float* __restrict__ maskf, float* __restrict__ scores) {
  int b = blockIdx.x, l0 = blockIdx.y * 64;
  int lane = threadIdx.x & 31, wave = threadIdx.x >> 5;
  const float* q = ctxq + (size_t)b * E;
  #pragma unroll 1
  for (int j = 0; j < 8; ++j) {
    int l = l0 + wave * 8 + j;
    const float* row = enc + ((size_t)b * L + l) * E;
    float s = 0.0f;
    for (int e = lane; e < E; e += 32) s += q[e] * row[e];
    s = wave_sum(s);
    if (lane == 0)
      scores[(size_t)b * L + l] = (maskf[(size_t)b * L + l] > 0.5f) ? s : NEGV;
  }
}

__global__ void k_stats(const float* __restrict__ scores, float* __restrict__ stats) {
  int b = blockIdx.x, tid = threadIdx.x;
  __shared__ float red[8];
  __shared__ float bmax;
  float mx = -3.0e38f;
  for (int l = tid; l < L; l += 256) mx = fmaxf(mx, scores[(size_t)b * L + l]);
  mx = wave_max(mx);
  if ((tid & 31) == 0) red[tid >> 5] = mx;
  __syncthreads();
  if (tid == 0) {
    float m = red[0];
    for (int i = 1; i < 8; ++i) m = fmaxf(m, red[i]);
    bmax = m;
  }
  __syncthreads();
  mx = bmax;
  float se = 0.0f;
  for (int l = tid; l < L; l += 256) se += __expf(scores[(size_t)b * L + l] - mx);
  se = wave_sum(se);
  __syncthreads();
  if ((tid & 31) == 0) red[tid >> 5] = se;
  __syncthreads();
  if (tid == 0) {
    float s = 0.0f;
    for (int i = 0; i < 8; ++i) s += red[i];
    stats[2 * b] = mx;
    stats[2 * b + 1] = s;
  }
}

__global__ void k_ctx_part(const float* __restrict__ scores, const float* __restrict__ stats,
                           const float* __restrict__ enc, float* __restrict__ part) {
  int b = blockIdx.x, ch = blockIdx.y, tid = threadIdx.x;   // 256 thr
  __shared__ float w[256];
  float mx = stats[2 * b], inv = 1.0f / stats[2 * b + 1];
  w[tid] = __expf(scores[(size_t)b * L + ch * 256 + tid] - mx) * inv;
  __syncthreads();
  float4 acc = make_float4(0.f, 0.f, 0.f, 0.f);
  const float* base = enc + ((size_t)b * L + ch * 256) * E + tid * 4;
  #pragma unroll 1
  for (int l = 0; l < 256; ++l) {
    float4 v = *(const float4*)(base + (size_t)l * E);
    float wl = w[l];
    acc.x += wl * v.x; acc.y += wl * v.y; acc.z += wl * v.z; acc.w += wl * v.w;
  }
  *(float4*)(part + ((size_t)(b * 8 + ch)) * E + tid * 4) = acc;
}

__global__ void k_ctx_reduce(const float* __restrict__ part, float* __restrict__ x) {
  int b = blockIdx.x, tid = threadIdx.x;   // 256 thr, 4 floats each
  float4 s = make_float4(0.f, 0.f, 0.f, 0.f);
  #pragma unroll
  for (int ch = 0; ch < 8; ++ch) {
    float4 v = *(const float4*)(part + ((size_t)(b * 8 + ch)) * E + tid * 4);
    s.x += v.x; s.y += v.y; s.z += v.z; s.w += v.w;
  }
  *(float4*)(x + (size_t)b * XK + EMB + tid * 4) = s;
}

__global__ void k_gru(const float* __restrict__ gi, const float* __restrict__ gh,
                      const float* __restrict__ hid, float* __restrict__ nh) {
  int b = blockIdx.x, tid = threadIdx.x;   // 256 thr, 4 d each
  #pragma unroll
  for (int i = 0; i < 4; ++i) {
    int d = tid * 4 + i;
    size_t o = (size_t)b * 3 * D;
    float ir = gi[o + d], iz = gi[o + D + d], in = gi[o + 2 * D + d];
    float hr = gh[o + d], hz = gh[o + D + d], hn = gh[o + 2 * D + d];
    float rg = 1.0f / (1.0f + __expf(-(ir + hr)));
    float zg = 1.0f / (1.0f + __expf(-(iz + hz)));
    float ng = tanhf(in + rg * hn);
    float hv = hid[(size_t)b * D + d];
    nh[(size_t)b * D + d] = (1.0f - zg) * ng + zg * hv;
  }
}

__global__ void k_span(const float* __restrict__ qs, const float* __restrict__ enc,
                       const float* __restrict__ maskf, float* __restrict__ o0,
                       float* __restrict__ o1, float* __restrict__ o2, float* __restrict__ o3) {
  int b = blockIdx.x, l0 = blockIdx.y * 64;
  int lane = threadIdx.x & 31, wave = threadIdx.x >> 5;
  const float* q0 = qs + 0 * (size_t)B * E + (size_t)b * E;
  const float* q1 = qs + 1 * (size_t)B * E + (size_t)b * E;
  const float* q2 = qs + 2 * (size_t)B * E + (size_t)b * E;
  const float* q3 = qs + 3 * (size_t)B * E + (size_t)b * E;
  #pragma unroll 1
  for (int j = 0; j < 8; ++j) {
    int l = l0 + wave * 8 + j;
    const float* row = enc + ((size_t)b * L + l) * E;
    float s0 = 0, s1 = 0, s2 = 0, s3 = 0;
    for (int e = lane; e < E; e += 32) {
      float x = row[e];
      s0 += q0[e] * x; s1 += q1[e] * x; s2 += q2[e] * x; s3 += q3[e] * x;
    }
    s0 = wave_sum(s0); s1 = wave_sum(s1); s2 = wave_sum(s2); s3 = wave_sum(s3);
    if (lane == 0) {
      bool ok = maskf[(size_t)b * L + l] > 0.5f;
      size_t idx = (size_t)b * L + l;
      o0[idx] = ok ? s0 : NEGV;
      o1[idx] = ok ? s1 : NEGV;
      o2[idx] = ok ? s2 : NEGV;
      o3[idx] = ok ? s3 : NEGV;
    }
  }
}

__global__ void k_cbias(const float* __restrict__ qref, const float* __restrict__ b0,
                        const float* __restrict__ b1, float* __restrict__ cb) {
  int b = blockIdx.x, hsel = blockIdx.y, tid = threadIdx.x;   // 256 thr
  const float* q = qref + (size_t)hsel * B * D + (size_t)b * D;
  const float* bias = hsel ? b1 : b0;
  __shared__ float red[8];
  float s = 0.0f;
  for (int e = tid; e < D; e += 256) s += q[e] * bias[e];
  s = wave_sum(s);
  if ((tid & 31) == 0) red[tid >> 5] = s;
  __syncthreads();
  if (tid == 0) {
    float t = 0.0f;
    for (int i = 0; i < 8; ++i) t += red[i];
    cb[hsel * B + b] = t;
  }
}

__global__ void k_ref(const float* __restrict__ qk, const float* __restrict__ cb,
                      const float* __restrict__ ph, const int* __restrict__ np,
                      float* __restrict__ r0, float* __restrict__ r1) {
  int b = blockIdx.x, s0i = blockIdx.y * 64;
  int lane = threadIdx.x & 31, wave = threadIdx.x >> 5;
  const float* q0 = qk + (size_t)b * D;
  const float* q1 = qk + (size_t)B * D + (size_t)b * D;
  int nvalid = np[b];
  #pragma unroll 1
  for (int j = 0; j < 8; ++j) {
    int s = s0i + wave * 8 + j;
    const float* row = ph + ((size_t)b * S + s) * D;
    float a0 = 0, a1 = 0;
    for (int e = lane; e < D; e += 32) {
      float x = row[e];
      a0 += q0[e] * x; a1 += q1[e] * x;
    }
    a0 = wave_sum(a0); a1 = wave_sum(a1);
    if (lane == 0) {
      bool ok = s < nvalid;
      r0[(size_t)b * S + s] = ok ? (a0 + cb[b]) : NEGV;
      r1[(size_t)b * S + s] = ok ? (a1 + cb[B + b]) : NEGV;
    }
  }
}

// ---------------- host side ----------------
static inline void launch_gemm(const float* A, const float* W, const float* bias,
                               float* C, int M, int N, int K, hipStream_t s) {
  dim3 g((M + 63) / 64, (N + 255) / 256);
  gemm_bf16<<<g, 256, 0, s>>>(A, W, bias, C, M, N, K);
}

extern "C" void kernel_launch(void* const* d_in, const int* in_sizes, int n_in,
                              void* d_out, int out_size, void* d_ws, size_t ws_size,
                              hipStream_t stream) {
  (void)in_sizes; (void)n_in; (void)out_size; (void)ws_size;
  const int*   prev_id = (const int*)d_in[0];
  const float* hidden  = (const float*)d_in[1];
  const float* enc     = (const float*)d_in[2];
  const float* maskf   = (const float*)d_in[3];
  const float* prevh   = (const float*)d_in[4];
  const int*   numprev = (const int*)d_in[5];
  const float* emb_w   = (const float*)d_in[6];
  const float* ctxq_w  = (const float*)d_in[7];
  const float* ctxq_b  = (const float*)d_in[8];
  const float* wih     = (const float*)d_in[9];
  const float* whh     = (const float*)d_in[10];
  const float* bih     = (const float*)d_in[11];
  const float* bhh     = (const float*)d_in[12];
  const float* op_w    = (const float*)d_in[13];
  const float* op_b    = (const float*)d_in[14];
  const float* a0_w    = (const float*)d_in[15];
  const float* a0_b    = (const float*)d_in[16];
  const float* a1_w    = (const float*)d_in[17];
  const float* a1_b    = (const float*)d_in[18];
  const float* span_w[4] = {(const float*)d_in[19], (const float*)d_in[21],
                            (const float*)d_in[23], (const float*)d_in[25]};
  const float* span_b[4] = {(const float*)d_in[20], (const float*)d_in[22],
                            (const float*)d_in[24], (const float*)d_in[26]};
  const float* r0q_w = (const float*)d_in[27];
  const float* r0q_b = (const float*)d_in[28];
  const float* r0k_w = (const float*)d_in[29];
  const float* r0k_b = (const float*)d_in[30];
  const float* r1q_w = (const float*)d_in[31];
  const float* r1q_b = (const float*)d_in[32];
  const float* r1k_w = (const float*)d_in[33];
  const float* r1k_b = (const float*)d_in[34];

  float* ws  = (float*)d_ws;
  float* out = (float*)d_out;

  float* x      = ws + WS_X;
  float* ctxq   = ws + WS_CTXQ;
  float* scores = ws + WS_SCORES;
  float* stats  = ws + WS_STATS;
  float* part   = ws + WS_PART;
  float* gi     = ws + WS_GI;
  float* gh     = ws + WS_GH;
  float* qspan  = ws + WS_QSPAN;
  float* qref   = ws + WS_QREF;
  float* qk     = ws + WS_QK;
  float* cb     = ws + WS_CB;
  float* wt0    = ws + WS_WT;                   // (D,E) = r0k_w^T
  float* wt1    = ws + WS_WT + (size_t)D * E;   // (D,E) = r1k_w^T

  float* o_hid = out + O_HID;

  // 1) embedding gather into x[:, 0:EMB]
  k_gather<<<dim3(B), EMB, 0, stream>>>(prev_id, emb_w, x);
  // transpose ref-key weights early (independent of everything else)
  k_transpose<<<dim3(D / 32, E / 32), 256, 0, stream>>>(r0k_w, wt0, E, D);
  k_transpose<<<dim3(D / 32, E / 32), 256, 0, stream>>>(r1k_w, wt1, E, D);
  // 2) ctx_query = hidden @ ctx_q_w^T + b
  launch_gemm(hidden, ctxq_w, ctxq_b, ctxq, B, E, D, stream);
  // 3) attention scores + softmax stats + context
  k_scores<<<dim3(B, L / 64), 256, 0, stream>>>(ctxq, enc, maskf, scores);
  k_stats<<<dim3(B), 256, 0, stream>>>(scores, stats);
  k_ctx_part<<<dim3(B, 8), 256, 0, stream>>>(scores, stats, enc, part);
  k_ctx_reduce<<<dim3(B), 256, 0, stream>>>(part, x);
  // 4) GRU gates
  launch_gemm(x, wih, bih, gi, B, 3 * D, XK, stream);
  launch_gemm(hidden, whh, bhh, gh, B, 3 * D, D, stream);
  k_gru<<<dim3(B), 256, 0, stream>>>(gi, gh, hidden, o_hid);
  // 5) classification heads
  launch_gemm(o_hid, op_w, op_b, out + O_OP, B, V, D, stream);
  launch_gemm(o_hid, a0_w, a0_b, out + O_A0, B, T, D, stream);
  launch_gemm(o_hid, a1_w, a1_b, out + O_A1, B, T, D, stream);
  // 6) span q-projections then one fused pass over encoder_out
  for (int hsel = 0; hsel < 4; ++hsel)
    launch_gemm(o_hid, span_w[hsel], span_b[hsel], qspan + (size_t)hsel * B * E,
                B, E, D, stream);
  k_span<<<dim3(B, L / 64), 256, 0, stream>>>(qspan, enc, maskf,
      out + O_S0S, out + O_S0E, out + O_S1S, out + O_S1E);
  // 7) ref heads via associativity: qk_h = (r{h}k_w)^T @ q_h, with W^T materialized
  launch_gemm(o_hid, r0q_w, r0q_b, qref, B, D, D, stream);
  launch_gemm(o_hid, r1q_w, r1q_b, qref + (size_t)B * D, B, D, D, stream);
  launch_gemm(qref, wt0, nullptr, qk, B, D, E, stream);
  launch_gemm(qref + (size_t)B * D, wt1, nullptr, qk + (size_t)B * D, B, D, E, stream);
  k_cbias<<<dim3(B, 2), 256, 0, stream>>>(qref, r0k_b, r1k_b, cb);
  k_ref<<<dim3(B, S / 64), 256, 0, stream>>>(qk, cb, prevh, numprev,
      out + O_R0, out + O_R1);
}
